// MultiHeadSelfAttention_29222957482013
// MI455X (gfx1250) — matrix operand order repaired and checked by reading, not compiled, not run on hardware
//
#include <hip/hip_runtime.h>
#include <hip/hip_bf16.h>

typedef unsigned short ushort_t;
typedef __bf16 bf16_t;
typedef bf16_t v16bf __attribute__((ext_vector_type(16)));
typedef float  v8f   __attribute__((ext_vector_type(8)));

#define BATCH 8
#define SLEN  1024
#define DMODEL 1024
#define NHEAD 16
#define HD    64

// ---------- helpers ----------

__device__ inline ushort_t f2bf(float f) {
    unsigned u = __float_as_uint(f);
    u += 0x7FFFu + ((u >> 16) & 1u);   // round-to-nearest-even
    return (ushort_t)(u >> 16);
}

union FragU {
    uint4  q[2];
    v16bf  v;
};

// A-matrix fragment (16x32 bf16), ISA 7.12.2 layout:
//   lanes 0-15 : M=lane,    elems 0-7 -> K=0..7,  elems 8-15 -> K=16..23
//   lanes 16-31: M=lane-16, elems 0-7 -> K=8..15, elems 8-15 -> K=24..31
__device__ inline v16bf loadA(const ushort_t* base, int ld, int lane) {
    const int m  = lane & 15;
    const int hb = (lane >> 4) * 8;
    const ushort_t* r = base + m * ld + hb;
    FragU u;
    u.q[0] = *(const uint4*)(r);
    u.q[1] = *(const uint4*)(r + 16);
    return u.v;
}

// B-matrix fragment (32x16 bf16) from an N-major source: B[k][n] = Src[n][k],
// Src row-major [N x K] with leading dim ld. lane = N (hi half K+8),
// elems 0-7 -> K=0..7, elems 8-15 -> K=16..23 (same as A).
__device__ inline v16bf loadBT(const ushort_t* base, int ld, int lane) {
    const int n  = lane & 15;
    const int kh = (lane >> 4) * 8;
    const ushort_t* r = base + n * ld + kh;
    FragU u;
    u.q[0] = *(const uint4*)(r);
    u.q[1] = *(const uint4*)(r + 16);
    return u.v;
}

__device__ inline v8f wmma_bf16(v16bf a, v16bf b, v8f c) {
    return __builtin_amdgcn_wmma_f32_16x16x32_bf16(
        false, a, false, b, (short)0, c, false, false);
}

// ---------- CDNA5 async copy: global -> LDS (ASYNCcnt tracked) ----------
__device__ inline void async_ld_b128(ushort_t* dst_lds, const ushort_t* src_global) {
    unsigned lds_off = (unsigned)(uintptr_t)dst_lds;
    asm volatile("global_load_async_to_lds_b128 %0, %1, off"
                 :: "v"(lds_off), "v"(src_global) : "memory");
}
#define ASYNC_WAIT(n) asm volatile("s_wait_asynccnt " #n ::: "memory")

// ---------- kernel 1: fp32 -> bf16 cast ----------

__global__ void cast_bf16_kernel(const float* __restrict__ in,
                                 ushort_t* __restrict__ out, int n) {
    int i = blockIdx.x * blockDim.x + threadIdx.x;
    if (i < n) out[i] = f2bf(in[i]);
}

// ---------- kernel 2: fused QKV projection ----------
// Per wave: 16(M) x 64(N) tile of Q, K, V. All 8 waves of a block share the
// same N-tile; the 3 weight tiles (64x32 bf16) are async-staged into LDS
// (double buffered). A fragment is preloaded one K-step ahead; B fragments
// are rotated one N-tile ahead so ds_loads overlap the WMMA chain.

__global__ __launch_bounds__(256) void qkv_kernel(
    const ushort_t* __restrict__ xb,
    const ushort_t* __restrict__ Wqb,
    const ushort_t* __restrict__ Wkb,
    const ushort_t* __restrict__ Wvb,
    const float* __restrict__ bq,
    const float* __restrict__ bk,
    const float* __restrict__ bv,
    ushort_t* __restrict__ Qg,
    ushort_t* __restrict__ Kg,
    ushort_t* __restrict__ Vtg) {

    __shared__ __align__(16) ushort_t Wbuf[3][2][64 * 32];  // 24 KB

    const int tid  = threadIdx.x;
    const int w    = tid >> 5;
    const int lane = tid & 31;
    const int gw   = blockIdx.x * 8 + w;      // 0..8191
    const int mt   = gw & 511;                // M-tile (differs per wave)
    const int nt4  = gw >> 9;                 // N-tile of 64 (same whole block)
    const int n0   = nt4 * 64;

    const ushort_t* Abase = xb + (size_t)mt * 16 * DMODEL;

    // staging map: 64 rows x 4 chunks of 8 halves = 256 lanes, 1 chunk/lane/matrix
    const int srow = tid >> 2;
    const int sc   = (tid & 3) * 8;
    const size_t wsrc = (size_t)(n0 + srow) * DMODEL + sc;

    auto stage = [&](int k0, int buf) {
        async_ld_b128(&Wbuf[0][buf][srow * 32 + sc], Wqb + wsrc + k0);
        async_ld_b128(&Wbuf[1][buf][srow * 32 + sc], Wkb + wsrc + k0);
        async_ld_b128(&Wbuf[2][buf][srow * 32 + sc], Wvb + wsrc + k0);
    };

    v8f aq[4], ak[4], av[4];
    v8f z = {};
    for (int t = 0; t < 4; ++t) { aq[t] = z; ak[t] = z; av[t] = z; }

    v16bf a = loadA(Abase, DMODEL, lane);     // K-chunk 0 preloaded
    stage(0, 0);
    for (int k0 = 0, it = 0; k0 < DMODEL; k0 += 32, ++it) {
        const int cur = it & 1;
        const bool more = (k0 + 32 < DMODEL);
        if (more) { stage(k0 + 32, cur ^ 1); ASYNC_WAIT(3); }
        else      { ASYNC_WAIT(0); }
        __syncthreads();                       // all waves' tiles visible

        // preload next A fragment (global) to overlap the WMMA chain
        v16bf anext = a;
        if (more) anext = loadA(Abase + k0 + 32, DMODEL, lane);

        // software-pipelined B fragments (distinct live ranges)
        v16bf fq = loadBT(&Wbuf[0][cur][0], 32, lane);
        v16bf fk = loadBT(&Wbuf[1][cur][0], 32, lane);
        v16bf fv = loadBT(&Wbuf[2][cur][0], 32, lane);
#pragma unroll
        for (int t = 0; t < 4; ++t) {
            v16bf nq = fq, nk = fk, nv = fv;
            if (t < 3) {
                const int toff = (t + 1) * 16 * 32;
                nq = loadBT(&Wbuf[0][cur][toff], 32, lane);
                nk = loadBT(&Wbuf[1][cur][toff], 32, lane);
                nv = loadBT(&Wbuf[2][cur][toff], 32, lane);
            }
            aq[t] = wmma_bf16(a, fq, aq[t]);
            ak[t] = wmma_bf16(a, fk, ak[t]);
            av[t] = wmma_bf16(a, fv, av[t]);
            fq = nq; fk = nk; fv = nv;
        }
        __syncthreads();                       // protect buffer being refilled
        a = anext;
    }

    const int nn = lane & 15;
    const int hi = lane >> 4;
    const int token0 = mt * 16;
    const int b  = token0 >> 10;
    const int s0 = token0 & 1023;
    const int h  = nt4;
    const float qscale = 0.125f;               // Hd^-0.5 folded into Q

    for (int t = 0; t < 4; ++t) {
        const int hd    = t * 16 + nn;
        const int nfeat = n0 + hd;
        const float bqv = bq[nfeat];
        const float bkv = bk[nfeat];
        const float bvv = bv[nfeat];
        for (int r = 0; r < 8; ++r) {
            const int s = s0 + r + hi * 8;
            const size_t qi = (((size_t)b * NHEAD + h) * SLEN + s) * HD + hd;
            Qg[qi] = f2bf((aq[t][r] + bqv) * qscale);
            Kg[qi] = f2bf(ak[t][r] + bkv);
            const size_t vi = (((size_t)b * NHEAD + h) * HD + hd) * SLEN + s;
            Vtg[vi] = f2bf(av[t][r] + bvv);
        }
    }
}

// ---------- kernel 3: flash attention ----------
// Per wave: one 16-row M-tile; all 8 waves of a block share (b,h), so K/V
// 32-key tiles are async-staged into LDS once per block (double buffered).

__global__ __launch_bounds__(256) void attn_kernel(
    const ushort_t* __restrict__ Qg,
    const ushort_t* __restrict__ Kg,
    const ushort_t* __restrict__ Vtg,
    float* __restrict__ out) {

    __shared__ __align__(16) ushort_t Kbuf[2][32 * 64];   // 8 KB
    __shared__ __align__(16) ushort_t Vbuf[2][64 * 32];   // 8 KB
    __shared__ __align__(16) ushort_t ldsP[8][16 * 32];   // 8 KB P staging

    const int tid  = threadIdx.x;
    const int w    = tid >> 5;
    const int lane = tid & 31;
    const int gw   = blockIdx.x * 8 + w;      // 0..8191
    const int mt   = gw & 63;                 // M-tile (differs per wave)
    const int bh   = gw >> 6;                 // same for the whole block

    const ushort_t* Qbase = Qg  + ((size_t)bh * SLEN + mt * 16) * HD;
    const ushort_t* Kbase = Kg  + (size_t)bh * SLEN * HD;
    const ushort_t* Vbase = Vtg + (size_t)bh * HD * SLEN;

    // staging maps: K tile 32 rows x 8 chunks; V tile 64 rows x 4 chunks
    const int krow = tid >> 3, kc = (tid & 7) * 8;
    const int vrow = tid >> 2, vc = (tid & 3) * 8;

    auto stage = [&](int kt, int buf) {
        async_ld_b128(&Kbuf[buf][krow * 64 + kc],
                      Kbase + (size_t)(kt * 32 + krow) * HD + kc);
        async_ld_b128(&Vbuf[buf][vrow * 32 + vc],
                      Vbase + (size_t)vrow * SLEN + kt * 32 + vc);
    };

    v16bf qf0 = loadA(Qbase, HD, lane);
    v16bf qf1 = loadA(Qbase + 32, HD, lane);

    v8f O[4];
    v8f z = {};
    for (int t = 0; t < 4; ++t) O[t] = z;
    float rm[8], rs[8];
    for (int r = 0; r < 8; ++r) { rm[r] = -__builtin_inff(); rs[r] = 0.f; }

    const int nn = lane & 15;
    const int hi = lane >> 4;
    ushort_t* pw = ldsP[w];

    stage(0, 0);
    for (int kt = 0; kt < SLEN / 32; ++kt) {
        const int cur = kt & 1;
        if (kt + 1 < SLEN / 32) { stage(kt + 1, cur ^ 1); ASYNC_WAIT(2); }
        else                    { ASYNC_WAIT(0); }
        __syncthreads();

        // scores: issue all 4 B fragments, then the 4 WMMAs
        const ushort_t* Kt = Kbuf[cur];                   // ld = 64
        v16bf kb0 = loadBT(Kt,                64, lane);
        v16bf kb1 = loadBT(Kt + 32,           64, lane);
        v16bf kb2 = loadBT(Kt + 16 * 64,      64, lane);
        v16bf kb3 = loadBT(Kt + 16 * 64 + 32, 64, lane);
        v8f s0 = z, s1 = z;
        s0 = wmma_bf16(qf0, kb0, s0);
        s0 = wmma_bf16(qf1, kb1, s0);
        s1 = wmma_bf16(qf0, kb2, s1);
        s1 = wmma_bf16(qf1, kb3, s1);

        // online softmax (scale already folded into Q)
        for (int r = 0; r < 8; ++r) {
            float mx = fmaxf(s0[r], s1[r]);
            for (int off = 8; off; off >>= 1)
                mx = fmaxf(mx, __shfl_xor(mx, off, 32));
            const float mnew = fmaxf(rm[r], mx);
            const float corr = __expf(rm[r] - mnew);
            const float p0 = __expf(s0[r] - mnew);
            const float p1 = __expf(s1[r] - mnew);
            float ps = p0 + p1;
            for (int off = 8; off; off >>= 1)
                ps += __shfl_xor(ps, off, 32);
            rs[r] = rs[r] * corr + ps;
            rm[r] = mnew;
            for (int t = 0; t < 4; ++t) O[t][r] *= corr;
            pw[(r + hi * 8) * 32 + nn]      = f2bf(p0);
            pw[(r + hi * 8) * 32 + 16 + nn] = f2bf(p1);
        }

        // O += P(16x32) x V(32x64): P read first (in-order LDS -> retires at
        // dscnt<=8), then the 4 V fragments, then the WMMAs.
        v16bf pA = loadA(pw, 32, lane);
        const ushort_t* Vt = Vbuf[cur];                   // ld = 32
        v16bf vb0 = loadBT(Vt,            32, lane);
        v16bf vb1 = loadBT(Vt + 1 * 512,  32, lane);
        v16bf vb2 = loadBT(Vt + 2 * 512,  32, lane);
        v16bf vb3 = loadBT(Vt + 3 * 512,  32, lane);
        O[0] = wmma_bf16(pA, vb0, O[0]);
        O[1] = wmma_bf16(pA, vb1, O[1]);
        O[2] = wmma_bf16(pA, vb2, O[2]);
        O[3] = wmma_bf16(pA, vb3, O[3]);

        __syncthreads();                       // protect buffer being refilled
    }

    const int b = bh >> 4;
    const int h = bh & 15;
    for (int t = 0; t < 4; ++t) {
        const int hd = t * 16 + nn;
        for (int r = 0; r < 8; ++r) {
            const int s = mt * 16 + r + hi * 8;
            out[(((size_t)b * SLEN + s) * NHEAD + h) * HD + hd] = O[t][r] / rs[r];
        }
    }
}

// ---------- launch ----------

extern "C" void kernel_launch(void* const* d_in, const int* in_sizes, int n_in,
                              void* d_out, int out_size, void* d_ws, size_t ws_size,
                              hipStream_t stream) {
    const float* x  = (const float*)d_in[0];
    const float* Wq = (const float*)d_in[1];
    const float* Wk = (const float*)d_in[2];
    const float* Wv = (const float*)d_in[3];
    const float* bq = (const float*)d_in[4];
    const float* bk = (const float*)d_in[5];
    const float* bv = (const float*)d_in[6];
    float* out = (float*)d_out;

    const size_t NX = (size_t)BATCH * SLEN * DMODEL;   // 8388608
    const size_t NW = (size_t)DMODEL * DMODEL;         // 1048576
    const size_t NQ = NX;

    ushort_t* p = (ushort_t*)d_ws;
    ushort_t* xb  = p; p += NX;
    ushort_t* Wqb = p; p += NW;
    ushort_t* Wkb = p; p += NW;
    ushort_t* Wvb = p; p += NW;
    ushort_t* Qg  = p; p += NQ;
    ushort_t* Kg  = p; p += NQ;
    ushort_t* Vtg = p; p += NQ;

    cast_bf16_kernel<<<(int)(NX / 256), 256, 0, stream>>>(x,  xb,  (int)NX);
    cast_bf16_kernel<<<(int)(NW / 256), 256, 0, stream>>>(Wq, Wqb, (int)NW);
    cast_bf16_kernel<<<(int)(NW / 256), 256, 0, stream>>>(Wk, Wkb, (int)NW);
    cast_bf16_kernel<<<(int)(NW / 256), 256, 0, stream>>>(Wv, Wvb, (int)NW);

    qkv_kernel<<<1024, 256, 0, stream>>>(xb, Wqb, Wkb, Wvb, bq, bk, bv, Qg, Kg, Vtg);
    attn_kernel<<<1024, 256, 0, stream>>>(Qg, Kg, Vtg, out);
}
